// ResidualVectorQuantizer_9629316677978
// MI455X (gfx1250) — compile-verified, matrix operand order checked by reading, not run on hardware
//
#include <hip/hip_runtime.h>

// ---------------------------------------------------------------------------
// Residual VQ (4 levels, 2048 codes, 128-dim, 32768 tokens) for MI455X/gfx1250
// Distance GEMM done with v_wmma_f32_16x16x32_f16 (f16 in, f32 acc): the
// scores only feed an argmin, so f16 inputs are fine; loss / x_q / residual
// are recomputed exactly in f32 from the gathered winning codes.
// Codebook tiles staged to LDS with CDNA5 async global->LDS copies.
// ---------------------------------------------------------------------------

typedef __attribute__((ext_vector_type(16))) _Float16 v16h;
typedef __attribute__((ext_vector_type(8)))  _Float16 v8h;
typedef __attribute__((ext_vector_type(4)))  _Float16 v4h;
typedef __attribute__((ext_vector_type(8)))  float    v8f;
typedef __attribute__((ext_vector_type(4)))  int      v4i;

#define NTOK 32768
#define NE   2048
#define ED   128
#define NQ   4
#define KC   128         // codes staged in LDS per block iteration (34 KB)
#define LROW 136         // padded LDS row (halves): 136*2=272B -> 4-bank skew

#if defined(__gfx1250__) && \
    __has_builtin(__builtin_amdgcn_global_load_async_to_lds_b128) && \
    __has_builtin(__builtin_amdgcn_s_wait_asynccnt)
#define RVQ_ASYNC_LDS 1
#else
#define RVQ_ASYNC_LDS 0
#endif

// builtin expects (addrspace(1) int4*, addrspace(3) int4*, imm, imm)
#define GPTR(p) ((__attribute__((address_space(1))) v4i*)(p))
#define LPTR(p) ((__attribute__((address_space(3))) v4i*)(p))

// ---------------------------------------------------------------------------
// Convert all 4 codebooks to f16 and compute ||c||^2 per code (once).
// One wave per codebook row.
// ---------------------------------------------------------------------------
__global__ __launch_bounds__(256)
void rvq_prep_cb(const float* __restrict__ cb_all,
                 _Float16* __restrict__ cb16,
                 float* __restrict__ cnorm)
{
    int gid  = blockIdx.x * 256 + threadIdx.x;
    int row  = gid >> 5;            // [0, NQ*NE)
    int lane = gid & 31;
    const float4 c = *(const float4*)(cb_all + (size_t)row * ED + lane * 4);
    v4h h; h[0] = (_Float16)c.x; h[1] = (_Float16)c.y;
           h[2] = (_Float16)c.z; h[3] = (_Float16)c.w;
    *(v4h*)(cb16 + (size_t)row * ED + lane * 4) = h;
    float ss = c.x*c.x + c.y*c.y + c.z*c.z + c.w*c.w;
    #pragma unroll
    for (int m = 1; m < 32; m <<= 1) ss += __shfl_xor(ss, m, 32);
    if (lane == 0) cnorm[row] = ss;
}

// ---------------------------------------------------------------------------
// residual <- x (f32 copy + f16 copy), zero loss accumulator.
// ---------------------------------------------------------------------------
__global__ __launch_bounds__(256)
void rvq_init(const float* __restrict__ x,
              float* __restrict__ res32,
              _Float16* __restrict__ res16,
              float* __restrict__ loss_acc)
{
    int gid = blockIdx.x * 256 + threadIdx.x;   // over NTOK*ED/4 float4 chunks
    const float4 v = *(const float4*)(x + (size_t)gid * 4);
    *(float4*)(res32 + (size_t)gid * 4) = v;
    v4h h; h[0] = (_Float16)v.x; h[1] = (_Float16)v.y;
           h[2] = (_Float16)v.z; h[3] = (_Float16)v.w;
    *(v4h*)(res16 + (size_t)gid * 4) = h;
    if (gid == 0) *loss_acc = 0.0f;
}

// ---------------------------------------------------------------------------
// WMMA distance GEMM + argmin for one level.
// Block: 128 threads = 4 wave32s; each wave owns 32 tokens (two 16-row tiles),
// block owns 128 tokens. Loop over all 2048 codes in LDS tiles of KC=128.
// Score s[n] = ||c_n||^2 - 2*(r . c_n)   (||r||^2 dropped: constant in argmin)
// ---------------------------------------------------------------------------
__global__ __launch_bounds__(128)
void rvq_argmin(const _Float16* __restrict__ res16,
                const _Float16* __restrict__ cb16,     // this level, [NE][ED]
                const float*    __restrict__ cnorm,    // this level, [NE]
                int*            __restrict__ idx_out)  // [NTOK]
{
    __shared__ _Float16 lds[KC * LROW];

    const int tid  = threadIdx.x;
    const int lane = tid & 31;
    const int wv   = tid >> 5;
    const int lm   = lane & 15;          // N (and M) index inside 16-tile
    const int hi   = lane >> 4;          // K-half selector for A/B fragments
    const int m0   = blockIdx.x * 128 + wv * 32;

    // ---- A fragments: 2 token tiles x 4 K-slices, resident all kernel ----
    v16h afrag[2][4];
    #pragma unroll
    for (int t = 0; t < 2; ++t) {
        const _Float16* rp = res16 + (size_t)(m0 + t * 16 + lm) * ED;
        #pragma unroll
        for (int s = 0; s < 4; ++s) {
            const int dbase = 32 * s + hi * 8;
            v8h lo = *(const v8h*)(rp + dbase);
            v8h hh = *(const v8h*)(rp + dbase + 16);
            afrag[t][s] = __builtin_shufflevector(lo, hh,
                0,1,2,3,4,5,6,7,8,9,10,11,12,13,14,15);
        }
    }

    float minval[2][8];
    int   minidx[2][8];
    #pragma unroll
    for (int t = 0; t < 2; ++t)
        #pragma unroll
        for (int r = 0; r < 8; ++r) { minval[t][r] = 3.4e38f; minidx[t][r] = 0; }

    for (int ct = 0; ct < NE / KC; ++ct) {
        __syncthreads();                 // previous tile fully consumed
        // cooperative stage: KC x ED halves -> padded LDS rows (16B chunks)
        const _Float16* cp = cb16 + (size_t)ct * KC * ED;
        #pragma unroll
        for (int i = 0; i < (KC * ED / 8) / 128; ++i) {
            int c   = tid + i * 128;
            int row = c >> 4;            // 16 eight-half chunks per row
            int ch  = c & 15;
#if RVQ_ASYNC_LDS
            __builtin_amdgcn_global_load_async_to_lds_b128(
                GPTR(cp + (size_t)row * ED + ch * 8),
                LPTR(&lds[row * LROW + ch * 8]), 0, 0);
#else
            *(v8h*)(&lds[row * LROW + ch * 8]) =
                *(const v8h*)(cp + (size_t)row * ED + ch * 8);
#endif
        }
#if RVQ_ASYNC_LDS
        __builtin_amdgcn_s_wait_asynccnt(0);
#endif
        __syncthreads();

        #pragma unroll
        for (int sub = 0; sub < KC / 16; ++sub) {
            const int nbase = ct * KC + sub * 16;
            const float cn  = cnorm[nbase + lm];
            v8f acc0 = {};
            v8f acc1 = {};
            #pragma unroll
            for (int s = 0; s < 4; ++s) {
                const int dbase = 32 * s + hi * 8;
                const _Float16* bp = &lds[(sub * 16 + lm) * LROW + dbase];
                v8h lo = *(const v8h*)bp;
                v8h hh = *(const v8h*)(bp + 16);
                v16h bfrag = __builtin_shufflevector(lo, hh,
                    0,1,2,3,4,5,6,7,8,9,10,11,12,13,14,15);
                acc0 = __builtin_amdgcn_wmma_f32_16x16x32_f16(
                    false, afrag[0][s], false, bfrag, (short)0, acc0, false, false);
                acc1 = __builtin_amdgcn_wmma_f32_16x16x32_f16(
                    false, afrag[1][s], false, bfrag, (short)0, acc1, false, false);
            }
            #pragma unroll
            for (int r = 0; r < 8; ++r) {
                float s0 = cn - 2.0f * acc0[r];
                if (s0 < minval[0][r]) { minval[0][r] = s0; minidx[0][r] = nbase + lm; }
                float s1 = cn - 2.0f * acc1[r];
                if (s1 < minval[1][r]) { minval[1][r] = s1; minidx[1][r] = nbase + lm; }
            }
        }
    }

    // ---- cross-lane (16-wide) lexicographic min-reduce; first-index ties ----
    #pragma unroll
    for (int t = 0; t < 2; ++t) {
        #pragma unroll
        for (int r = 0; r < 8; ++r) {
            float v = minval[t][r];
            int   i = minidx[t][r];
            #pragma unroll
            for (int m = 1; m <= 8; m <<= 1) {
                float ov = __shfl_xor(v, m, 32);
                int   oi = __shfl_xor(i, m, 32);
                if (ov < v || (ov == v && oi < i)) { v = ov; i = oi; }
            }
            if (lm == 0) {
                int tok = m0 + t * 16 + r + hi * 8;   // D layout: M = r + 8*(lane/16)
                idx_out[tok] = i;
            }
        }
    }
}

// ---------------------------------------------------------------------------
// Exact f32 update for one level: gather winning code, residual -= q,
// x_q += q, loss += sum((q - r)^2), emit index. One wave per token.
// ---------------------------------------------------------------------------
__global__ __launch_bounds__(256)
void rvq_update(const float* __restrict__ cb,        // this level f32 [NE][ED]
                const int*   __restrict__ idx,
                float*       __restrict__ res32,
                _Float16*    __restrict__ res16,
                float*       __restrict__ xq,        // d_out[0 .. NTOK*ED)
                float*       __restrict__ out_idx,   // d_out idx region
                float*       __restrict__ loss_acc,
                int level, int first)
{
    int gid  = blockIdx.x * 256 + threadIdx.x;
    int tok  = gid >> 5;
    int lane = gid & 31;
    size_t off = (size_t)tok * ED + lane * 4;

    int k = idx[tok];
    const float4 q = *(const float4*)(cb + (size_t)k * ED + lane * 4);
    float4 r = *(const float4*)(res32 + off);

    float4 d; d.x = q.x - r.x; d.y = q.y - r.y; d.z = q.z - r.z; d.w = q.w - r.w;
    float ss = d.x*d.x + d.y*d.y + d.z*d.z + d.w*d.w;

    float4 rn; rn.x = -d.x; rn.y = -d.y; rn.z = -d.z; rn.w = -d.w;   // r - q
    *(float4*)(res32 + off) = rn;
    v4h h; h[0] = (_Float16)rn.x; h[1] = (_Float16)rn.y;
           h[2] = (_Float16)rn.z; h[3] = (_Float16)rn.w;
    *(v4h*)(res16 + off) = h;

    float4 xo;
    if (first) { xo = q; }
    else {
        xo = *(const float4*)(xq + off);
        xo.x += q.x; xo.y += q.y; xo.z += q.z; xo.w += q.w;
    }
    *(float4*)(xq + off) = xo;

    #pragma unroll
    for (int m = 1; m < 32; m <<= 1) ss += __shfl_xor(ss, m, 32);
    if (lane == 0) {
        atomicAdd(loss_acc, ss);
        out_idx[(size_t)tok * NQ + level] = (float)k;
    }
}

// mean_loss = mean_l [ (1+beta) * sumsq_l / (NTOK*ED) ] = 1.25*total/(4*N*D)
__global__ void rvq_finish(const float* __restrict__ loss_acc,
                           float* __restrict__ out_loss)
{
    *out_loss = 1.25f * (*loss_acc) / (4.0f * (float)NTOK * (float)ED);
}

// ---------------------------------------------------------------------------
extern "C" void kernel_launch(void* const* d_in, const int* in_sizes, int n_in,
                              void* d_out, int out_size, void* d_ws, size_t ws_size,
                              hipStream_t stream)
{
    (void)in_sizes; (void)n_in; (void)out_size; (void)ws_size;

    const float* x      = (const float*)d_in[0];   // [NTOK][ED]
    const float* cb_all = (const float*)d_in[1];   // [NQ][NE][ED]

    float* out_xq   = (float*)d_out;                       // NTOK*ED
    float* out_loss = (float*)d_out + (size_t)NTOK * ED;   // 1
    float* out_idx  = out_loss + 1;                        // NTOK*NQ

    // workspace carve-up
    char* ws = (char*)d_ws;
    float*    res32    = (float*)ws;                               // 16 MB
    _Float16* res16    = (_Float16*)(ws + (size_t)NTOK*ED*4);      //  8 MB
    _Float16* cb16     = (_Float16*)(ws + (size_t)NTOK*ED*6);      //  2 MB
    float*    cnorm    = (float*)(ws + (size_t)NTOK*ED*6 + (size_t)NQ*NE*ED*2);
    int*      idxbuf   = (int*)((char*)cnorm + (size_t)NQ*NE*4);
    float*    loss_acc = (float*)((char*)idxbuf + (size_t)NTOK*4);

    rvq_prep_cb<<<(NQ * NE * 32) / 256, 256, 0, stream>>>(cb_all, cb16, cnorm);
    rvq_init<<<(NTOK * ED / 4) / 256, 256, 0, stream>>>(x, res32, res16, loss_acc);

    for (int l = 0; l < NQ; ++l) {
        const _Float16* cb16_l  = cb16  + (size_t)l * NE * ED;
        const float*    cnorm_l = cnorm + (size_t)l * NE;
        const float*    cb_l    = cb_all + (size_t)l * NE * ED;

        rvq_argmin<<<NTOK / 128, 128, 0, stream>>>(res16, cb16_l, cnorm_l, idxbuf);
        rvq_update<<<(NTOK * 32) / 256, 256, 0, stream>>>(
            cb_l, idxbuf, res32, res16, out_xq, out_idx, loss_acc, l, l == 0);
    }

    rvq_finish<<<1, 1, 0, stream>>>(loss_acc, out_loss);
}
